// AutoWeightedCELoss_54236847014020
// MI455X (gfx1250) — 8-bit matrix operand order repaired and checked by reading, compiled, not run on hardware
//
#include <hip/hip_runtime.h>
#include <hip/hip_bf16.h>
#include <stdint.h>


#define B_N 8
#define C_N 4
#define H_N 512
#define W_N 512
#define NPIX (B_N * H_N * W_N)
#define NBLK (B_N * (H_N / 32) * (W_N / 32))

typedef __attribute__((ext_vector_type(8))) int v8i;

__device__ __forceinline__ uint32_t lds_off(const void* p) {
  return (uint32_t)(uintptr_t)p;
}
__device__ __forceinline__ void async_ld_b8(uint32_t lds, const void* g) {
  asm volatile("global_load_async_to_lds_b8 %0, %1, off" :: "v"(lds), "v"(g) : "memory");
}
__device__ __forceinline__ void async_ld_b128(uint32_t lds, const void* g) {
  asm volatile("global_load_async_to_lds_b128 %0, %1, off" :: "v"(lds), "v"(g) : "memory");
}

__global__ void awce_zero(int* hist) {
  if (threadIdx.x < 4) hist[threadIdx.x] = 0;
}

__global__ __launch_bounds__(256) void awce_hist(const int* __restrict__ lab,
                                                 int* __restrict__ hist) {
  __shared__ int h[4];
  if (threadIdx.x < 4) h[threadIdx.x] = 0;
  __syncthreads();
  int idx = blockIdx.x * 256 + threadIdx.x;
  int stride = gridDim.x * 256;
  int c0 = 0, c1 = 0, c2 = 0, c3 = 0;
  for (int i = idx; i < NPIX; i += stride) {
    int l = lab[i];
    l = min(max(l, 0), 3);
    c0 += (l == 0); c1 += (l == 1); c2 += (l == 2); c3 += (l == 3);
  }
  atomicAdd(&h[0], c0); atomicAdd(&h[1], c1);
  atomicAdd(&h[2], c2); atomicAdd(&h[3], c3);
  __syncthreads();
  if (threadIdx.x < 4) atomicAdd(&hist[threadIdx.x], h[threadIdx.x]);
}

__global__ __launch_bounds__(128) void awce_main(const float* __restrict__ cls,
                                                 const int* __restrict__ lab,
                                                 const int* __restrict__ hist,
                                                 float* __restrict__ partials) {
  __shared__ __align__(16) unsigned char sLab[64 * 64];
  __shared__ __align__(16) unsigned char sCol[5 * 4 * 32 * 64];
  __shared__ __align__(16) float sCls[4 * 32 * 32];
  __shared__ float sCW[4];
  __shared__ float sWsum[4];

  const int t = threadIdx.x;
  const int b = blockIdx.z;
  const int tileH = blockIdx.y * 32;
  const int tileW = blockIdx.x * 32;

#pragma unroll 4
  for (int i = 0; i < 32; ++i) {
    int q = i * 128 + t;
    int lr = q >> 6, lc = q & 63;
    int h = tileH - 16 + lr, w = tileW - 16 + lc;
    int hc = min(max(h, 0), H_N - 1), wc = min(max(w, 0), W_N - 1);
    const unsigned char* g =
        (const unsigned char*)(lab + (((size_t)b << 18) + ((size_t)hc << 9) + wc));
    async_ld_b8(lds_off(&sLab[q]), g);
  }
#pragma unroll
  for (int i = 0; i < 8; ++i) {
    int q = i * 128 + t;
    int c = q >> 8, rem = q & 255, r = rem >> 3, wg = (rem & 7) << 2;
    const float* g = cls + ((((size_t)b * 4 + c) << 18) +
                            ((size_t)(tileH + r) << 9) + (size_t)(tileW + wg));
    async_ld_b128(lds_off(&sCls[(c * 32 + r) * 32 + wg]), g);
  }
  if (t < 4) {
    float freq = (float)hist[t] / (float)NPIX;
    sCW[t] = 2.0f / (freq + 1.0f);
  }
  asm volatile("s_wait_asynccnt 8" ::: "memory");
  __syncthreads();

  for (int i = 0; i < 32; ++i) {
    int q = i * 128 + t;
    int lr = q >> 6, lc = q & 63;
    int h = tileH - 16 + lr, w = tileW - 16 + lc;
    if (h < 0 || h >= H_N || w < 0 || w >= W_N) sLab[q] = 0xFF;
  }
  __syncthreads();

  for (int i = 0; i < 16; ++i) {
    int q = i * 128 + t;
    int r = q >> 6, w = q & 63;
    int base = (16 + r) * 64 + w;
    uint32_t cnt = 0;
#define ADDROW(dh) { unsigned char L = sLab[base + (dh) * 64]; \
                     if (L < 4u) cnt += (1u << (L * 8)); }
#define STORE5(s) { int off = ((s) * 4) * 2048 + r * 64 + w;                 \
      sCol[off]        = (unsigned char)(cnt);                               \
      sCol[off + 2048] = (unsigned char)(cnt >> 8);                          \
      sCol[off + 4096] = (unsigned char)(cnt >> 16);                         \
      sCol[off + 6144] = (unsigned char)(cnt >> 24); }
    ADDROW(-1) ADDROW(0) ADDROW(1) STORE5(0)
    ADDROW(-2) ADDROW(2)           STORE5(1)
    ADDROW(-3) ADDROW(3) ADDROW(-4) ADDROW(4) STORE5(2)
#pragma unroll
    for (int d = 5; d <= 8; ++d) { ADDROW(-d) ADDROW(d) }
    STORE5(3)
#pragma unroll
    for (int d = 9; d <= 16; ++d) { ADDROW(-d) ADDROW(d) }
    STORE5(4)
#undef ADDROW
#undef STORE5
  }
  asm volatile("s_wait_asynccnt 0" ::: "memory");
  __syncthreads();

  const int wv = t >> 5, lane = t & 31, laneLo = lane & 15, laneHi = lane >> 4;
  const int sbr = wv >> 1, sbc = wv & 1;
  const int Arow = sbr * 16 + laneLo;

  int kbA[8];
  {
    const int t0[8] = {0, 4, 16, 20, 32, 36, 48, 52};
#pragma unroll
    for (int v = 0; v < 8; ++v) kbA[v] = t0[v] + laneHi * 8;
  }
  int lbl8[8];
#pragma unroll
  for (int r = 0; r < 8; ++r) {
    int pr = sbr * 16 + laneHi * 8 + r, pc = sbc * 16 + laneLo;
    lbl8[r] = sLab[(pr + 16) * 64 + (pc + 16)];
  }
  float pixw[8];
#pragma unroll
  for (int r = 0; r < 8; ++r) pixw[r] = 1.0f;

  const int c0 = 16 * (sbc + 1);
#pragma unroll
  for (int s = 0; s < 5; ++s) {
    const int p = 1 << s;
    v8i bmat;
#pragma unroll
    for (int v = 0; v < 8; ++v) {
      int Kb = kbA[v];
      uint32_t word = 0;
#pragma unroll
      for (int by = 0; by < 4; ++by) {
        int d = (Kb + by) - laneLo - c0;
        word |= ((d >= -p) && (d <= p)) ? (1u << (by * 8)) : 0u;
      }
      bmat[v] = (int)word;
    }
    v8i dch[4];
#pragma unroll
    for (int c = 0; c < 4; ++c) {
      v8i a;
      const unsigned char* plane = &sCol[(s * 4 + c) * 2048 + Arow * 64];
#pragma unroll
      for (int v = 0; v < 8; ++v) a[v] = *(const int*)(plane + kbA[v]);
      v8i cz = {0, 0, 0, 0, 0, 0, 0, 0};
      dch[c] = __builtin_amdgcn_wmma_i32_16x16x64_iu8(false, a, false, bmat,
                                                      cz, false, false);
    }
    const int kk = 2 * p + 1;
    const float wgt = 1.0f / (float)(kk * kk - 1);
#pragma unroll
    for (int r = 0; r < 8; ++r) {
      int pr = sbr * 16 + laneHi * 8 + r, pc = sbc * 16 + laneLo;
      int h = tileH + pr, w = tileW + pc;
      int lbl = lbl8[r];
      int mth = (lbl & 1) ? ((lbl & 2) ? dch[3][r] : dch[1][r])
                          : ((lbl & 2) ? dch[2][r] : dch[0][r]);
      int rows = min(h + p, H_N - 1) - max(h - p, 0) + 1;
      int colsn = min(w + p, W_N - 1) - max(w - p, 0) + 1;
      pixw[r] += wgt * (float)(rows * colsn - mth);
    }
  }

  float lsum = 0.0f;
#pragma unroll
  for (int r = 0; r < 8; ++r) {
    int pr = sbr * 16 + laneHi * 8 + r, pc = sbc * 16 + laneLo;
    int lbl = lbl8[r];
    float x0 = sCls[(0 * 32 + pr) * 32 + pc];
    float x1 = sCls[(1 * 32 + pr) * 32 + pc];
    float x2 = sCls[(2 * 32 + pr) * 32 + pc];
    float x3 = sCls[(3 * 32 + pr) * 32 + pc];
    float mx = fmaxf(fmaxf(x0, x1), fmaxf(x2, x3));
    float se = expf(x0 - mx) + expf(x1 - mx) + expf(x2 - mx) + expf(x3 - mx);
    float lse = mx + logf(se);
    float xl = (lbl & 1) ? ((lbl & 2) ? x3 : x1) : ((lbl & 2) ? x2 : x0);
    lsum += sCW[lbl] * (lse - xl) * pixw[r];
  }
#pragma unroll
  for (int off = 16; off >= 1; off >>= 1) lsum += __shfl_xor(lsum, off, 32);
  if (lane == 0) sWsum[wv] = lsum;
  __syncthreads();
  if (t == 0) {
    int bid = (blockIdx.z * 16 + blockIdx.y) * 16 + blockIdx.x;
    partials[bid] = sWsum[0] + sWsum[1] + sWsum[2] + sWsum[3];
  }
}

__global__ void awce_final(const float* __restrict__ partials,
                           float* __restrict__ out) {
  float acc = 0.0f;
  for (int i = threadIdx.x; i < NBLK; i += 32) acc += partials[i];
#pragma unroll
  for (int off = 16; off >= 1; off >>= 1) acc += __shfl_xor(acc, off, 32);
  if (threadIdx.x == 0) out[0] = acc * (1.0f / (float)NPIX);
}

extern "C" void kernel_launch(void* const* d_in, const int* in_sizes, int n_in,
                              void* d_out, int out_size, void* d_ws, size_t ws_size,
                              hipStream_t stream) {
  (void)in_sizes; (void)n_in; (void)out_size; (void)ws_size;
  const float* cls = (const float*)d_in[0];
  const int* lab = (const int*)d_in[1];
  int* hist = (int*)d_ws;
  float* partials = (float*)((char*)d_ws + 16);
  float* out = (float*)d_out;

  awce_zero<<<1, 32, 0, stream>>>(hist);
  awce_hist<<<512, 256, 0, stream>>>(lab, hist);
  dim3 grid(W_N / 32, H_N / 32, B_N);
  awce_main<<<grid, 128, 0, stream>>>(cls, lab, hist, partials);
  awce_final<<<1, 32, 0, stream>>>(partials, out);
}
